// AnswerModule_20916490731648
// MI455X (gfx1250) — matrix operand order repaired and checked by reading, not compiled, not run on hardware
//
#include <hip/hip_runtime.h>

// ---------------- constants ----------------
#define BB 64          // batch
#define HH 1024        // hidden
#define VV 50257       // vocab
#define TT 20          // steps
#define N3 3072        // 3*H
#define KX 2048        // 2*H
#define NT 4           // N-tiles per wave (64 columns) -> 4x A-fragment reuse

typedef __attribute__((ext_vector_type(16))) __bf16 v16bf;
typedef __attribute__((ext_vector_type(8)))  __bf16 v8bf;
typedef __attribute__((ext_vector_type(8)))  float  v8f;

// ---------------- fragment loaders (CDNA5 wave32 WMMA; this code's contiguous k order, same in A and B) ----------------
// A: 16x32 bf16.  lane&15 = M row; lanes 0-15 hold K=0..15, lanes 16-31 K=16..31
__device__ __forceinline__ v16bf load_a_frag(const __bf16* __restrict__ A, int lda,
                                             int m0, int k0, int lane) {
    int m  = m0 + (lane & 15);
    int kh = (lane >> 4) * 16;
    const __bf16* p = A + (long)m * lda + k0 + kh;
    v8bf c0 = *(const v8bf*)(p);
    v8bf c1 = *(const v8bf*)(p + 8);
    v16bf a;
#pragma unroll
    for (int i = 0; i < 8; ++i) { a[i] = c0[i]; a[i + 8] = c1[i]; }
    return a;
}

// B: 32x16 bf16.  lane&15 = N col; lanes 0-15 hold K=0..15, lanes 16-31 K=16..31.
// Column n of B == row n of the (N x K) row-major weight matrix -> contiguous load.
// Row pointer is pre-clamped (out-of-range columns never consumed) -> branch-free K-loop.
__device__ __forceinline__ v16bf load_b_frag_bf16(const __bf16* __restrict__ row,
                                                  int k0, int lane) {
    return *(const v16bf*)(row + k0 + (lane >> 4) * 16);
}

__device__ __forceinline__ v16bf load_b_frag_f32(const float* __restrict__ row,
                                                 int k0, int lane) {
    const float* p = row + k0 + (lane >> 4) * 16;
    v16bf b;
#pragma unroll
    for (int i = 0; i < 16; ++i) b[i] = (__bf16)p[i];
    return b;
}

// ---------------- generic 64xN = 64xK * (NxK)^T WMMA kernel ----------------
// Each wave: 4 M-tiles x NT N-tiles (64x64 output), 16 WMMAs per 32-K chunk.
// __launch_bounds__(128, 1): allow the full VGPR budget (acc tile = 128 VGPRs) -> no spills.
// ARGMAX=true : fuse +bias, per-row argmax over N via packed-key atomicMax (keys zeroed by init).
// ARGMAX=false: store C = acc + bias to Cout (row-major 64 x N).
// WBF16 selects the weight element type at compile time.
template <bool ARGMAX, bool WBF16>
__global__ void __launch_bounds__(128, 1)
gemm64_wmma_kernel(const __bf16* __restrict__ A,
                   const void*   __restrict__ Wv,
                   const float*  __restrict__ bias,
                   int N, int K,
                   float* __restrict__ Cout,
                   unsigned long long* __restrict__ keyOut) {
    const int lane  = threadIdx.x & 31;
    const int wave  = threadIdx.x >> 5;
    const int n0    = (blockIdx.x * (blockDim.x >> 5) + wave) * (NT * 16);
    if (n0 >= N) return;  // uniform per-wave exit (before any WMMA)

    int nj[NT], njc[NT];
    const __bf16* wrow_b[NT];
    const float*  wrow_f[NT];
#pragma unroll
    for (int j = 0; j < NT; ++j) {
        nj[j]  = n0 + j * 16 + (lane & 15);
        njc[j] = (nj[j] < N) ? nj[j] : (N - 1);  // clamp: garbage for n>=N never consumed
        if (WBF16) wrow_b[j] = (const __bf16*)Wv + (long)njc[j] * K;
        else       wrow_f[j] = (const float*) Wv + (long)njc[j] * K;
    }

    v8f acc[4][NT];
#pragma unroll
    for (int t = 0; t < 4; ++t)
#pragma unroll
        for (int j = 0; j < NT; ++j)
#pragma unroll
            for (int i = 0; i < 8; ++i) acc[t][j][i] = 0.0f;

    for (int k0 = 0; k0 < K; k0 += 32) {
        v16bf bfrag[NT];
#pragma unroll
        for (int j = 0; j < NT; ++j)
            bfrag[j] = WBF16 ? load_b_frag_bf16(wrow_b[j], k0, lane)
                             : load_b_frag_f32 (wrow_f[j], k0, lane);
#pragma unroll
        for (int t = 0; t < 4; ++t) {
            v16bf afrag = load_a_frag(A, K, t * 16, k0, lane);
#pragma unroll
            for (int j = 0; j < NT; ++j)
                acc[t][j] = __builtin_amdgcn_wmma_f32_16x16x32_bf16(
                    false, afrag, false, bfrag[j], (short)0, acc[t][j], false, false);
        }
    }

    // C/D layout: lane&15 = N col, rows m = t*16 + (lane>>4)*8 + r
    if (ARGMAX) {
        float    bval[NT];
        unsigned ninv[NT];
#pragma unroll
        for (int j = 0; j < NT; ++j) {
            bval[j] = bias[njc[j]];
            ninv[j] = 0xFFFFFFFFu - (unsigned)nj[j];  // smaller index wins ties under max
        }
#pragma unroll
        for (int t = 0; t < 4; ++t) {
#pragma unroll
            for (int r = 0; r < 8; ++r) {
                // local max across this lane's NT columns first
                unsigned long long key = 0ull;
#pragma unroll
                for (int j = 0; j < NT; ++j) {
                    float val = acc[t][j][r] + bval[j];
                    unsigned u = __float_as_uint(val);
                    u = (u & 0x80000000u) ? ~u : (u | 0x80000000u);  // order-preserving map
                    unsigned long long kj = (nj[j] < N)
                        ? (((unsigned long long)u << 32) | (unsigned long long)ninv[j]) : 0ull;
                    key = (kj > key) ? kj : key;
                }
                // reduce across the 16 lanes holding this row's columns
#pragma unroll
                for (int m = 1; m < 16; m <<= 1) {
                    unsigned long long o = __shfl_xor(key, m);
                    key = (o > key) ? o : key;
                }
                if ((lane & 15) == 0) {
                    int row = t * 16 + (lane >> 4) * 8 + r;
                    atomicMax(&keyOut[row], key);
                }
            }
        }
    } else {
#pragma unroll
        for (int j = 0; j < NT; ++j) {
            if (nj[j] < N) {
                const float bval = bias[nj[j]];
#pragma unroll
                for (int t = 0; t < 4; ++t)
#pragma unroll
                    for (int r = 0; r < 8; ++r) {
                        int row = t * 16 + (lane >> 4) * 8 + r;
                        Cout[(long)row * N + nj[j]] = acc[t][j][r] + bval;
                    }
            }
        }
    }
}

// ---------------- helper kernels ----------------
__global__ void f32_to_bf16_kernel(const float* __restrict__ src, __bf16* __restrict__ dst, long n) {
    long i = (long)blockIdx.x * blockDim.x + threadIdx.x;
    long stride = (long)gridDim.x * blockDim.x;
    for (; i < n; i += stride) dst[i] = (__bf16)src[i];
}

// h = M; h_bf16 = bf16(M); x[:,1024:2048] = bf16(q); zero argmax keys (all 20 steps)
__global__ void init_kernel(const float* __restrict__ M, const float* __restrict__ q,
                            float* __restrict__ h, __bf16* __restrict__ hb,
                            __bf16* __restrict__ x, unsigned long long* __restrict__ keys) {
    int i = blockIdx.x * blockDim.x + threadIdx.x;  // 65536 threads
    int b = i >> 10, k = i & 1023;
    float m = M[i];
    h[i]  = m;
    hb[i] = (__bf16)m;
    x[(long)b * KX + HH + k] = (__bf16)q[i];
    if (i < TT * BB) keys[i] = 0ull;
}

// decode argmax key -> token; write token to output; gather embedding row (fp32->bf16) into x[:, :1024]
__global__ void gather_kernel(const unsigned long long* __restrict__ keys,
                              const float* __restrict__ emb,
                              __bf16* __restrict__ x, int* __restrict__ out_tok, int step) {
    int b = blockIdx.x;
    unsigned long long key = keys[b];
    int v = (int)(0xFFFFFFFFu - (unsigned)(key & 0xFFFFFFFFull));
    if (threadIdx.x == 0) out_tok[b * TT + step] = v;
    const float* row = emb + (long)v * HH;
    for (int k = threadIdx.x; k < HH; k += blockDim.x)
        x[(long)b * KX + k] = (__bf16)row[k];
}

// GRU gate math: r,z,n -> h_new (fp32 state + bf16 copy for next step's WMMA)
__global__ void gru_gate_kernel(const float* __restrict__ gi, const float* __restrict__ gh,
                                float* __restrict__ h, __bf16* __restrict__ hb) {
    int i = blockIdx.x * blockDim.x + threadIdx.x;  // 65536 threads
    int b = i >> 10, j = i & 1023;
    const float* gib = gi + (long)b * N3;
    const float* ghb = gh + (long)b * N3;
    float ir = gib[j], iz = gib[HH + j], in_ = gib[2 * HH + j];
    float hr = ghb[j], hz = ghb[HH + j], hn = ghb[2 * HH + j];
    float r = 1.0f / (1.0f + __expf(-(ir + hr)));
    float z = 1.0f / (1.0f + __expf(-(iz + hz)));
    float nn = tanhf(in_ + r * hn);
    float hnew = (1.0f - z) * nn + z * h[i];
    h[i]  = hnew;
    hb[i] = (__bf16)hnew;
}

// ---------------- host launcher ----------------
extern "C" void kernel_launch(void* const* d_in, const int* in_sizes, int n_in,
                              void* d_out, int out_size, void* d_ws, size_t ws_size,
                              hipStream_t stream) {
    const float* M        = (const float*)d_in[0];
    const float* questions= (const float*)d_in[1];
    const float* word_emb = (const float*)d_in[2];
    const float* W_out    = (const float*)d_in[3];
    const float* b_out    = (const float*)d_in[4];
    const float* W_ih     = (const float*)d_in[5];
    const float* W_hh     = (const float*)d_in[6];
    const float* b_ih     = (const float*)d_in[7];
    const float* b_hh     = (const float*)d_in[8];
    int* out_tok = (int*)d_out;

    char* p = (char*)d_ws;
    auto alloc = [&](size_t bytes) -> char* {
        char* r = p;
        p += (bytes + 255) & ~(size_t)255;
        return r;
    };
    float*  h_f32  = (float*)              alloc((size_t)BB * HH * 4);
    __bf16* h_bf16 = (__bf16*)             alloc((size_t)BB * HH * 2);
    __bf16* x_bf16 = (__bf16*)             alloc((size_t)BB * KX * 2);
    float*  gi     = (float*)              alloc((size_t)BB * N3 * 4);
    float*  gh     = (float*)              alloc((size_t)BB * N3 * 4);
    unsigned long long* keys = (unsigned long long*)alloc((size_t)TT * BB * 8);
    __bf16* Wih_b  = (__bf16*)             alloc((size_t)N3 * KX * 2);
    __bf16* Whh_b  = (__bf16*)             alloc((size_t)N3 * HH * 2);
    size_t used_small = (size_t)(p - (char*)d_ws);
    // L2-resident bf16 copy of W_out (103 MB fits 192 MB L2) if workspace allows
    __bf16* Wout_b = nullptr;
    size_t wout_bytes = (size_t)VV * HH * 2;
    if (ws_size >= used_small + wout_bytes + 256)
        Wout_b = (__bf16*)alloc(wout_bytes);

    // ---- per-launch prep ----
    f32_to_bf16_kernel<<<4096, 256, 0, stream>>>(W_ih, Wih_b, (long)N3 * KX);
    f32_to_bf16_kernel<<<4096, 256, 0, stream>>>(W_hh, Whh_b, (long)N3 * HH);
    if (Wout_b)
        f32_to_bf16_kernel<<<8192, 256, 0, stream>>>(W_out, Wout_b, (long)VV * HH);
    init_kernel<<<256, 256, 0, stream>>>(M, questions, h_f32, h_bf16, x_bf16, keys);

    const int BLK = 128;  // 4 waves per block, NT n-tiles per wave
    const int ntilesV = (VV + 15) / 16;                    // 3142
    const int gridV   = (ntilesV + 4 * NT - 1) / (4 * NT); // 197
    const int gridG   = (N3 / 16 + 4 * NT - 1) / (4 * NT); // 12

    for (int t = 0; t < TT; ++t) {
        // logits = h @ W_out^T + b_out, fused argmax
        if (Wout_b)
            gemm64_wmma_kernel<true, true><<<gridV, BLK, 0, stream>>>(
                h_bf16, Wout_b, b_out, VV, HH, nullptr, keys + (size_t)t * BB);
        else
            gemm64_wmma_kernel<true, false><<<gridV, BLK, 0, stream>>>(
                h_bf16, W_out, b_out, VV, HH, nullptr, keys + (size_t)t * BB);
        // x = [emb[tok], q]; emit token
        gather_kernel<<<BB, 256, 0, stream>>>(keys + (size_t)t * BB, word_emb, x_bf16, out_tok, t);
        // gi = x @ W_ih^T + b_ih   (K = 2048)
        gemm64_wmma_kernel<false, true><<<gridG, BLK, 0, stream>>>(
            x_bf16, Wih_b, b_ih, N3, KX, gi, nullptr);
        // gh = h @ W_hh^T + b_hh   (K = 1024)
        gemm64_wmma_kernel<false, true><<<gridG, BLK, 0, stream>>>(
            h_bf16, Whh_b, b_hh, N3, HH, gh, nullptr);
        // gates -> h_new
        gru_gate_kernel<<<256, 256, 0, stream>>>(gi, gh, h_f32, h_bf16);
    }
}
